// VideoAttention_6777458394023
// MI455X (gfx1250) — matrix operand order repaired and checked by reading, not compiled, not run on hardware
//
#include <hip/hip_runtime.h>
#include <hip/hip_bf16.h>
#include <math.h>

typedef __attribute__((ext_vector_type(16))) __bf16        v16bf;
typedef __attribute__((ext_vector_type(8)))  float         v8f;
typedef __attribute__((ext_vector_type(8)))  unsigned int  v8u;

#define WMMA_BF16(a, b, c) \
    __builtin_amdgcn_wmma_f32_16x16x32_bf16(false, (a), false, (b), (short)0, (c), false, false)

// ---- problem constants (match setup_inputs) ----
#define BATCH   2
#define TFRAMES 8
#define NIMG    (BATCH * TFRAMES)      // 16
#define CCH     768
#define HW      256
#define HEADS   12
#define HDIM    64                     // C / HEADS
#define SEQ     (TFRAMES * HW)         // 2048
#define BM      (BATCH * HEADS)        // 24
#define C3      (3 * CCH)              // 2304

// ---------------------------------------------------------------------------
// A-fragment (16x32 bf16, M x K): a[e] = row[16*g + e]
// loaded as 8 dwords from a row-major row (pairs contiguous in K).
// ---------------------------------------------------------------------------
__device__ __forceinline__ v16bf frag_a_row(const __bf16* row, int g) {
    v8u u;
    const unsigned int* p = (const unsigned int*)row;
#pragma unroll
    for (int r = 0; r < 8; ++r) {
        int e2 = 16 * g + (r << 1);  // K index of element 2r
        u[r] = p[e2 >> 1];
    }
    return __builtin_bit_cast(v16bf, u);
}

// ---------------------------------------------------------------------------
// B-fragment (32x16 bf16, K x N): b[e] = B[16*g + e][N].
// Stored N-major (Bt[N][K], K contiguous) -> one contiguous 32B read.
// ---------------------------------------------------------------------------
__device__ __forceinline__ v16bf frag_b_col(const __bf16* rowN, int g) {
    v8u u;
    const unsigned int* p = (const unsigned int*)(rowN + 16 * g);
#pragma unroll
    for (int r = 0; r < 8; ++r) u[r] = p[r];
    return __builtin_bit_cast(v16bf, u);
}

// ===========================================================================
// K0: EDM2 mp_weight normalization -> bf16.  One block per output row.
// rows [0, 2304): w_qkv ; rows [2304, 3072): w_proj
// ===========================================================================
__global__ __launch_bounds__(256) void k_wnorm(const float* __restrict__ wqkv,
                                               const float* __restrict__ wproj,
                                               __bf16* __restrict__ wnq,
                                               __bf16* __restrict__ wnp) {
    int row = blockIdx.x;
    const float* src;
    __bf16* dst;
    if (row < C3) { src = wqkv + (size_t)row * CCH;        dst = wnq + (size_t)row * CCH; }
    else          { src = wproj + (size_t)(row - C3) * CCH; dst = wnp + (size_t)(row - C3) * CCH; }

    __shared__ float red[256];
    float s = 0.f;
    for (int i = threadIdx.x; i < CCH; i += 256) { float v = src[i]; s += v * v; }
    red[threadIdx.x] = s;
    __syncthreads();
#pragma unroll
    for (int st = 128; st > 0; st >>= 1) {
        if (threadIdx.x < st) red[threadIdx.x] += red[threadIdx.x + st];
        __syncthreads();
    }
    float norm  = sqrtf(red[0]);
    float scale = (1.0f / sqrtf((float)CCH)) / (1e-4f + norm * sqrtf(1.0f / (float)CCH));
    for (int i = threadIdx.x; i < CCH; i += 256) dst[i] = (__bf16)(src[i] * scale);
}

// ===========================================================================
// K1: pack x [img][ch][pix] f32 -> Xt [img][pix][ch] bf16
// ===========================================================================
__global__ __launch_bounds__(256) void k_pack_x(const float* __restrict__ x,
                                                __bf16* __restrict__ xt) {
    int idx = blockIdx.x * 256 + threadIdx.x;
    if (idx >= NIMG * CCH * HW) return;
    int pix = idx & (HW - 1);
    int ch  = (idx >> 8) % CCH;
    int img = idx / (CCH * HW);
    xt[((size_t)img * HW + pix) * CCH + ch] = (__bf16)x[idx];
}

// ===========================================================================
// K2: QKV GEMM.  QKVbuf[img][2304][256] = Wn[2304x768] x Xt[img]^T
// One wave = 16x64 output tile; 8 waves / block.
// ===========================================================================
__global__ __launch_bounds__(256) void k_qkv_gemm(const __bf16* __restrict__ wn,
                                                  const __bf16* __restrict__ xt,
                                                  __bf16* __restrict__ qkv) {
    int wtile = blockIdx.x * 8 + (threadIdx.x >> 5);       // 16*144*4 = 9216 tiles
    int lane  = threadIdx.x & 31;
    int g     = lane >> 4;
    int ln    = lane & 15;
    int img = wtile / (144 * 4);
    int rem = wtile % (144 * 4);
    int mt  = rem >> 2;                                    // 144 M-tiles
    int nt  = rem & 3;                                     // 4 N-tiles of 64 pixels

    const __bf16* arow  = wn + (size_t)(mt * 16 + ln) * CCH;
    const __bf16* bbase = xt + ((size_t)img * HW + nt * 64) * CCH;

    v8f acc[4] = {};
#pragma unroll 4
    for (int kb = 0; kb < CCH; kb += 32) {
        v16bf a = frag_a_row(arow + kb, g);
#pragma unroll
        for (int n = 0; n < 4; ++n) {
            v16bf b = frag_b_col(bbase + (size_t)(n * 16 + ln) * CCH + kb, g);
            acc[n] = WMMA_BF16(a, b, acc[n]);
        }
    }
    __bf16* out = qkv + (size_t)img * C3 * HW;
#pragma unroll
    for (int n = 0; n < 4; ++n)
#pragma unroll
        for (int v = 0; v < 8; ++v) {
            int ch  = mt * 16 + v + 8 * g;
            int pix = nt * 64 + n * 16 + ln;
            out[(size_t)ch * HW + pix] = (__bf16)acc[n][v];
        }
}

// ===========================================================================
// K3: RoPE + repack.  Q,K: [bm][S][64] bf16 (rotated); Vt: [bm][64][S] bf16
// one thread per (bm, s, j<32)
// ===========================================================================
__global__ __launch_bounds__(256) void k_rope(const __bf16* __restrict__ qkv,
                                              __bf16* __restrict__ Q,
                                              __bf16* __restrict__ K,
                                              __bf16* __restrict__ Vt) {
    int idx = blockIdx.x * 256 + threadIdx.x;
    if (idx >= BM * SEQ * 32) return;
    int j  = idx & 31;
    int s  = (idx >> 5) & (SEQ - 1);
    int bm = idx >> 16;                                    // SEQ*32 = 65536
    int b_idx = bm / HEADS, head = bm % HEADS;
    int t_idx = s >> 8, pix = s & (HW - 1);
    int img   = b_idx * TFRAMES + t_idx;

    const __bf16* base = qkv + (size_t)img * C3 * HW;
    int chq = head * HDIM + j;
    float q1 = (float)base[(size_t)(chq) * HW + pix];
    float q2 = (float)base[(size_t)(chq + 32) * HW + pix];
    float k1 = (float)base[(size_t)(CCH + chq) * HW + pix];
    float k2 = (float)base[(size_t)(CCH + chq + 32) * HW + pix];
    float v1 = (float)base[(size_t)(2 * CCH + chq) * HW + pix];
    float v2 = (float)base[(size_t)(2 * CCH + chq + 32) * HW + pix];

    // inv_freq = 10000^(-2j/64);  ang = t * inv_freq
    float inv_freq = __expf(-((float)(2 * j) / 64.0f) * 9.210340371976184f);
    float ang = (float)t_idx * inv_freq;
    float cs = __cosf(ang), sn = __sinf(ang);

    size_t ro = ((size_t)bm * SEQ + s) * HDIM;
    Q[ro + j]      = (__bf16)(q1 * cs - q2 * sn);
    Q[ro + j + 32] = (__bf16)(q1 * sn + q2 * cs);
    K[ro + j]      = (__bf16)(k1 * cs - k2 * sn);
    K[ro + j + 32] = (__bf16)(k1 * sn + k2 * cs);
    Vt[((size_t)bm * HDIM + j) * SEQ + s]      = (__bf16)v1;
    Vt[((size_t)bm * HDIM + j + 32) * SEQ + s] = (__bf16)v2;
}

// ===========================================================================
// K4: block-causal flash attention.
// grid: bm(24) x qchunk(32); block = 4 waves; wave = one 16-row q-tile.
// k-blocks of 64; causal limit is frame-granular so no partial masking.
// Output written to Yt[img][pix][ch=cdim*12+head] bf16.
// ===========================================================================
__global__ __launch_bounds__(128) void k_attn(const __bf16* __restrict__ Q,
                                              const __bf16* __restrict__ K,
                                              const __bf16* __restrict__ Vt,
                                              __bf16* __restrict__ Yt) {
    __shared__ __bf16 plds[4][16 * 64];                    // per-wave P tile
    int wid  = threadIdx.x >> 5;
    int lane = threadIdx.x & 31;
    int g    = lane >> 4;
    int ln   = lane & 15;
    int bm    = blockIdx.x >> 5;                           // 24
    int chunk = blockIdx.x & 31;                           // 32 chunks of 64 rows
    int q0    = chunk * 64 + wid * 16;
    int b_idx = bm / HEADS, head = bm % HEADS;
    int k_end = ((q0 >> 8) + 1) << 8;                      // (frame+1)*256

    const __bf16* qb = Q  + (size_t)bm * SEQ * HDIM;
    const __bf16* kb_ = K + (size_t)bm * SEQ * HDIM;
    const __bf16* vb = Vt + (size_t)bm * HDIM * SEQ;

    v16bf aq0 = frag_a_row(qb + (size_t)(q0 + ln) * HDIM, g);
    v16bf aq1 = frag_a_row(qb + (size_t)(q0 + ln) * HDIM + 32, g);

    v8f o[4] = {};
    float m_run[8], l_run[8];
#pragma unroll
    for (int v = 0; v < 8; ++v) { m_run[v] = -1e30f; l_run[v] = 0.f; }

    for (int kblk = 0; kblk < k_end; kblk += 64) {
        // ---- scores: 4 tiles of 16x16, K-loop over head dim 64 ----
        v8f s[4];
#pragma unroll
        for (int n = 0; n < 4; ++n) {
            const __bf16* krow = kb_ + (size_t)(kblk + n * 16 + ln) * HDIM;
            v8f sc = {};
            sc = WMMA_BF16(aq0, frag_b_col(krow, g), sc);
            sc = WMMA_BF16(aq1, frag_b_col(krow + 32, g), sc);
            s[n] = sc * 0.125f;                            // 1/sqrt(64)
        }
        // ---- online softmax row stats (rows live across 16-lane groups) ----
        float alpha[8], rs[8];
#pragma unroll
        for (int v = 0; v < 8; ++v) {
            float mt = fmaxf(fmaxf(s[0][v], s[1][v]), fmaxf(s[2][v], s[3][v]));
#pragma unroll
            for (int off = 1; off < 16; off <<= 1) mt = fmaxf(mt, __shfl_xor(mt, off, 32));
            float mnew = fmaxf(m_run[v], mt);
            alpha[v]   = __expf(m_run[v] - mnew);
            m_run[v]   = mnew;
            rs[v]      = 0.f;
        }
#pragma unroll
        for (int n = 0; n < 4; ++n)
#pragma unroll
            for (int v = 0; v < 8; ++v) {
                float p = __expf(s[n][v] - m_run[v]);
                s[n][v] = p;
                rs[v] += p;
            }
#pragma unroll
        for (int v = 0; v < 8; ++v) {
#pragma unroll
            for (int off = 1; off < 16; off <<= 1) rs[v] += __shfl_xor(rs[v], off, 32);
            l_run[v] = l_run[v] * alpha[v] + rs[v];
        }
#pragma unroll
        for (int n = 0; n < 4; ++n)
#pragma unroll
            for (int v = 0; v < 8; ++v) o[n][v] *= alpha[v];

        // ---- transpose P through private LDS slice (D-layout -> A-layout) ----
        __bf16* pw = plds[wid];
#pragma unroll
        for (int n = 0; n < 4; ++n)
#pragma unroll
            for (int v = 0; v < 8; ++v)
                pw[(v + 8 * g) * 64 + n * 16 + ln] = (__bf16)s[n][v];
        asm volatile("s_wait_dscnt 0" ::: "memory");       // same-wave LDS RAW

        v16bf ap0 = frag_a_row(pw + ln * 64, g);
        v16bf ap1 = frag_a_row(pw + ln * 64 + 32, g);

        // ---- O += P x V  (V read transposed: Vt[dim][token]) ----
#pragma unroll
        for (int nd = 0; nd < 4; ++nd) {
            const __bf16* vrow = vb + (size_t)(nd * 16 + ln) * SEQ + kblk;
            o[nd] = WMMA_BF16(ap0, frag_b_col(vrow, g), o[nd]);
            o[nd] = WMMA_BF16(ap1, frag_b_col(vrow + 32, g), o[nd]);
        }
    }

    // ---- finalize + scatter into Yt[img][pix][cdim*12+head] ----
    int t_idx = q0 >> 8;
    int img   = b_idx * TFRAMES + t_idx;
    __bf16* yb = Yt + (size_t)img * HW * CCH;
#pragma unroll
    for (int v = 0; v < 8; ++v) {
        float inv_l = 1.0f / l_run[v];
        int pix = (q0 + v + 8 * g) & (HW - 1);
#pragma unroll
        for (int nd = 0; nd < 4; ++nd) {
            int ch = (nd * 16 + ln) * HEADS + head;
            yb[(size_t)pix * CCH + ch] = (__bf16)(o[nd][v] * inv_l);
        }
    }
}

// ===========================================================================
// K5: output projection + mp_sum residual.  out[img][768][256] f32
// ===========================================================================
__global__ __launch_bounds__(256) void k_proj(const __bf16* __restrict__ wp,
                                              const __bf16* __restrict__ yt,
                                              const float* __restrict__ x,
                                              float* __restrict__ out) {
    int wtile = blockIdx.x * 8 + (threadIdx.x >> 5);       // 16*48*4 = 3072 tiles
    int lane  = threadIdx.x & 31;
    int g     = lane >> 4;
    int ln    = lane & 15;
    int img = wtile / (48 * 4);
    int rem = wtile % (48 * 4);
    int mt  = rem >> 2;
    int nt  = rem & 3;

    const __bf16* arow  = wp + (size_t)(mt * 16 + ln) * CCH;
    const __bf16* bbase = yt + ((size_t)img * HW + nt * 64) * CCH;

    v8f acc[4] = {};
#pragma unroll 4
    for (int kbk = 0; kbk < CCH; kbk += 32) {
        v16bf a = frag_a_row(arow + kbk, g);
#pragma unroll
        for (int n = 0; n < 4; ++n) {
            v16bf b = frag_b_col(bbase + (size_t)(n * 16 + ln) * CCH + kbk, g);
            acc[n] = WMMA_BF16(a, b, acc[n]);
        }
    }
    // mp_sum: (x*(1-tb) + yo*tb) / sqrt((1-tb)^2 + tb^2),  tb = 0.3
    const float wa = 0.7f * 1.3130643285972254f;           // (1-tb)*scale
    const float wb = 0.3f * 1.3130643285972254f;           // tb*scale
    const float* xi = x   + (size_t)img * CCH * HW;
    float*       oo = out + (size_t)img * CCH * HW;
#pragma unroll
    for (int n = 0; n < 4; ++n)
#pragma unroll
        for (int v = 0; v < 8; ++v) {
            int ch  = mt * 16 + v + 8 * g;
            int pix = nt * 64 + n * 16 + ln;
            size_t off = (size_t)ch * HW + pix;
            oo[off] = xi[off] * wa + acc[n][v] * wb;
        }
}

// ===========================================================================
// host launcher
// ===========================================================================
extern "C" void kernel_launch(void* const* d_in, const int* in_sizes, int n_in,
                              void* d_out, int out_size, void* d_ws, size_t ws_size,
                              hipStream_t stream) {
    const float* x      = (const float*)d_in[0];
    const float* w_qkv  = (const float*)d_in[1];
    const float* w_proj = (const float*)d_in[2];
    float* out = (float*)d_out;

    char* ws = (char*)d_ws;
    size_t off = 0;
    auto take = [&](size_t bytes) { char* p = ws + off; off += (bytes + 255) & ~(size_t)255; return p; };

    __bf16* wnq  = (__bf16*)take((size_t)C3 * CCH * 2);
    __bf16* wnp  = (__bf16*)take((size_t)CCH * CCH * 2);
    __bf16* xt   = (__bf16*)take((size_t)NIMG * HW * CCH * 2);
    __bf16* qkv  = (__bf16*)take((size_t)NIMG * C3 * HW * 2);
    __bf16* Qb   = (__bf16*)take((size_t)BM * SEQ * HDIM * 2);
    __bf16* Kb   = (__bf16*)take((size_t)BM * SEQ * HDIM * 2);
    __bf16* Vtb  = (__bf16*)take((size_t)BM * HDIM * SEQ * 2);
    __bf16* Yt   = (__bf16*)take((size_t)NIMG * HW * CCH * 2);

    k_wnorm<<<C3 + CCH, 256, 0, stream>>>(w_qkv, w_proj, wnq, wnp);

    int nx = NIMG * CCH * HW;
    k_pack_x<<<(nx + 255) / 256, 256, 0, stream>>>(x, xt);

    k_qkv_gemm<<<(16 * 144 * 4) / 8, 256, 0, stream>>>(wnq, xt, qkv);

    int nr = BM * SEQ * 32;
    k_rope<<<(nr + 255) / 256, 256, 0, stream>>>(qkv, Qb, Kb, Vtb);

    k_attn<<<BM * 32, 128, 0, stream>>>(Qb, Kb, Vtb, Yt);

    k_proj<<<(16 * 48 * 4) / 8, 256, 0, stream>>>(wnp, Yt, x, out);
}
